// GCNModel_61349312856088
// MI455X (gfx1250) — compile-verified, matrix operand order checked by reading, not run on hardware
//
#include <hip/hip_runtime.h>
#include <hip/hip_bf16.h>

// ---------------------------------------------------------------------------
// GCN pipeline for MI455X (gfx1250, wave32, WMMA).
//   x1 = sigmoid(Agg(X @ W1) + b1)
//   x2 = sigmoid(Agg(x1 @ W2) + b2)
//   y  = relu(x2 @ Wm1 + bm1) @ Wm2 + bm2
// GEMMs: v_wmma_f32_16x16x32_f16 with an explicit ping-pong register pipeline
// (two fragment sets, manually 2x-unrolled K loop) -> no rotation movs, loads
// for the next K-chunk overlap WMMAs of the current one.
// Edge aggregation: float4 gather + global_atomic_add_f32.
// ---------------------------------------------------------------------------

typedef __attribute__((ext_vector_type(16))) _Float16 v16h;
typedef __attribute__((ext_vector_type(8)))  _Float16 v8h;
typedef __attribute__((ext_vector_type(8)))  float    v8f;

// Per-lane 16-bit A/B fragment (16x32 MxK): lane = {r15 = lane&15, half = lane>>4}
// v[0..7]  = K[8*half .. +7], v[8..15] = K[16+8*half .. +7].
// Caller pre-offsets the row pointer by (8*half) halfs; both loads 16B aligned.
__device__ __forceinline__ v16h ld_frag(const _Float16* __restrict__ p) {
  v8h lo = *(const v8h*)(p);
  v8h hi = *(const v8h*)(p + 16);
  v16h r;
#pragma unroll
  for (int i = 0; i < 8; ++i) { r[i] = lo[i]; r[i + 8] = hi[i]; }
  return r;
}

// D = A(MxK,f16) * Bt(NxK,f16)^T, f32 acc.  K/32 must be even (K in {256,512}).
// Wave tile: (MT*16) x (NT*16). Block: 4 waves stacked in M -> 4*MT*16 rows.
// mode: 0 = store f32, 1 = f16 relu(x+bias), 2 = f32 x+bias, col guard n<Nreal.
template <int MT, int NT>
__global__ __launch_bounds__(128) void k_gemm_wmma(
    const _Float16* __restrict__ A, const _Float16* __restrict__ Bt,
    float* __restrict__ Cf, _Float16* __restrict__ Ch,
    const float* __restrict__ bias,
    int M, int K, int ldC, int Nreal, int mode)
{
  const int lane = threadIdx.x & 31;
  const int wave = threadIdx.x >> 5;
  const int m0   = (blockIdx.x * 4 + wave) * (MT * 16);
  if (m0 >= M) return;                       // wave-uniform guard
  const int n0   = blockIdx.y * (16 * NT);
  const int r15  = lane & 15;
  const int koff = (lane >> 4) << 3;         // 0 or 8 halfs

  // A row pointers (row-clamped so the partial last block never reads OOB)
  const _Float16* __restrict__ ap[MT];
#pragma unroll
  for (int mi = 0; mi < MT; ++mi) {
    int row = m0 + mi * 16 + r15;
    row = row < M ? row : (M - 1);
    ap[mi] = A + (size_t)row * K + koff;
  }
  // B row pointers (Bt is padded to the launched column count)
  const _Float16* __restrict__ bp[NT];
#pragma unroll
  for (int t = 0; t < NT; ++t)
    bp[t] = Bt + (size_t)(n0 + t * 16 + r15) * K + koff;

  const v8f vzero = {0.f, 0.f, 0.f, 0.f, 0.f, 0.f, 0.f, 0.f};
  v8f acc[MT][NT];
#pragma unroll
  for (int mi = 0; mi < MT; ++mi)
#pragma unroll
    for (int t = 0; t < NT; ++t) acc[mi][t] = vzero;

  // ---- ping-pong pipeline over NC = K/32 chunks (NC even) ----
  const int NC = K >> 5;
  v16h a0[MT], b0[NT], a1[MT], b1[NT];

  // prologue: chunk 0 -> set0
#pragma unroll
  for (int mi = 0; mi < MT; ++mi) a0[mi] = ld_frag(ap[mi]);
#pragma unroll
  for (int t = 0; t < NT; ++t)    b0[t]  = ld_frag(bp[t]);

  for (int c = 1; c + 1 < NC; c += 2) {
    const int k1 = c << 5;
#pragma unroll
    for (int mi = 0; mi < MT; ++mi) {
      a1[mi] = ld_frag(ap[mi] + k1);
      __builtin_prefetch(ap[mi] + k1 + 256, 0, 3);   // global_prefetch_b8, A stream
    }
#pragma unroll
    for (int t = 0; t < NT; ++t) b1[t] = ld_frag(bp[t] + k1);

#pragma unroll
    for (int mi = 0; mi < MT; ++mi)
#pragma unroll
      for (int t = 0; t < NT; ++t)
        acc[mi][t] = __builtin_amdgcn_wmma_f32_16x16x32_f16(
            false, a0[mi], false, b0[t], (short)0, acc[mi][t], false, false);

    const int k2 = k1 + 32;
#pragma unroll
    for (int mi = 0; mi < MT; ++mi) a0[mi] = ld_frag(ap[mi] + k2);
#pragma unroll
    for (int t = 0; t < NT; ++t)    b0[t]  = ld_frag(bp[t] + k2);

#pragma unroll
    for (int mi = 0; mi < MT; ++mi)
#pragma unroll
      for (int t = 0; t < NT; ++t)
        acc[mi][t] = __builtin_amdgcn_wmma_f32_16x16x32_f16(
            false, a1[mi], false, b1[t], (short)0, acc[mi][t], false, false);
  }

  // tail: set0 holds chunk NC-2; load chunk NC-1 into set1, drain both
  {
    const int kl = (NC - 1) << 5;
#pragma unroll
    for (int mi = 0; mi < MT; ++mi) a1[mi] = ld_frag(ap[mi] + kl);
#pragma unroll
    for (int t = 0; t < NT; ++t)    b1[t]  = ld_frag(bp[t] + kl);

#pragma unroll
    for (int mi = 0; mi < MT; ++mi)
#pragma unroll
      for (int t = 0; t < NT; ++t)
        acc[mi][t] = __builtin_amdgcn_wmma_f32_16x16x32_f16(
            false, a0[mi], false, b0[t], (short)0, acc[mi][t], false, false);
#pragma unroll
    for (int mi = 0; mi < MT; ++mi)
#pragma unroll
      for (int t = 0; t < NT; ++t)
        acc[mi][t] = __builtin_amdgcn_wmma_f32_16x16x32_f16(
            false, a1[mi], false, b1[t], (short)0, acc[mi][t], false, false);
  }

  // C/D layout: VGPR r -> row base + r + 8*(lane>=16), col n0 + (lane&15)
  const int half8 = (lane >> 4) << 3;
#pragma unroll
  for (int mi = 0; mi < MT; ++mi) {
    const int rbase = m0 + mi * 16 + half8;
#pragma unroll
    for (int t = 0; t < NT; ++t) {
      const int n = n0 + t * 16 + r15;
      if (mode == 2 && n >= Nreal) continue;
      const float bv = (mode == 0) ? 0.f : bias[n];
#pragma unroll
      for (int r = 0; r < 8; ++r) {
        const int row = rbase + r;
        if (row >= M) continue;
        const size_t idx = (size_t)row * (size_t)ldC + n;
        const float v = acc[mi][t][r] + bv;
        if (mode == 1) Ch[idx] = (_Float16)fmaxf(v, 0.f);
        else           Cf[idx] = v;
      }
    }
  }
}

// ---------------- elementwise / prep kernels ----------------

__global__ void k_fill(float* __restrict__ p, float v, int n) {
  int i = blockIdx.x * blockDim.x + threadIdx.x;
  if (i < n) p[i] = v;
}

__global__ void k_deg_edges(const int* __restrict__ dst, float* __restrict__ deg, int E) {
  int e = blockIdx.x * blockDim.x + threadIdx.x;
  if (e < E) unsafeAtomicAdd(&deg[dst[e]], 1.0f);
}

__global__ void k_dinv(float* __restrict__ d, int n) {
  int i = blockIdx.x * blockDim.x + threadIdx.x;
  if (i < n) d[i] = rsqrtf(fmaxf(d[i], 1.0f));
}

__global__ void k_f32_to_f16(const float* __restrict__ in, _Float16* __restrict__ out, size_t n) {
  const size_t stride = (size_t)gridDim.x * blockDim.x;
  for (size_t i = (size_t)blockIdx.x * blockDim.x + threadIdx.x; i < n; i += stride)
    out[i] = (_Float16)in[i];
}

// Wt[n*K + k] = (n < Nc) ? W[k*Nc + n] : 0   for n in [0, Npad)
__global__ void k_transpose_w(const float* __restrict__ W, _Float16* __restrict__ Wt,
                              int K, int Nc, int Npad) {
  const size_t total  = (size_t)Npad * K;
  const size_t stride = (size_t)gridDim.x * blockDim.x;
  for (size_t i = (size_t)blockIdx.x * blockDim.x + threadIdx.x; i < total; i += stride) {
    const int n = (int)(i / K), k = (int)(i % K);
    Wt[i] = (n < Nc) ? (_Float16)W[(size_t)k * Nc + n] : (_Float16)0.f;
  }
}

// agg[i,f] = H[i,f] * dinv[i]^2 + bias[f]   (self-loop contribution + bias)
__global__ void k_agg_init(const float* __restrict__ H, const float* __restrict__ dinv,
                           const float* __restrict__ bias, float* __restrict__ agg,
                           size_t total, int F) {
  const size_t stride = (size_t)gridDim.x * blockDim.x;
  for (size_t i = (size_t)blockIdx.x * blockDim.x + threadIdx.x; i < total; i += stride) {
    const float di = dinv[i / F];
    agg[i] = H[i] * di * di + bias[i % F];
  }
}

// agg[dst,:] += H[src,:] * dinv[src]*dinv[dst]
// One block per edge; blockDim.x == F/4; float4 gather + 4x global_atomic_add_f32.
__global__ void k_agg_edges(const int* __restrict__ src, const int* __restrict__ dst,
                            const float* __restrict__ dinv, const float* __restrict__ H,
                            float* __restrict__ agg) {
  const int e = blockIdx.x;
  const int s = src[e], d = dst[e];
  const float w = dinv[s] * dinv[d];
  const int F = blockDim.x * 4;
  const float4* __restrict__ hs = (const float4*)(H + (size_t)s * F);
  float* __restrict__ ad = agg + (size_t)d * F;
  const int f = threadIdx.x;
  const float4 v = hs[f];
  unsafeAtomicAdd(&ad[f * 4 + 0], v.x * w);
  unsafeAtomicAdd(&ad[f * 4 + 1], v.y * w);
  unsafeAtomicAdd(&ad[f * 4 + 2], v.z * w);
  unsafeAtomicAdd(&ad[f * 4 + 3], v.w * w);
}

__global__ void k_sigmoid_f16(const float* __restrict__ in, _Float16* __restrict__ out, size_t n) {
  const size_t stride = (size_t)gridDim.x * blockDim.x;
  for (size_t i = (size_t)blockIdx.x * blockDim.x + threadIdx.x; i < n; i += stride)
    out[i] = (_Float16)(1.0f / (1.0f + __expf(-in[i])));
}

// ---------------- launcher ----------------

extern "C" void kernel_launch(void* const* d_in, const int* in_sizes, int n_in,
                              void* d_out, int out_size, void* d_ws, size_t ws_size,
                              hipStream_t stream)
{
  const float* features = (const float*)d_in[0];
  const int*   edges    = (const int*)  d_in[1];   // int32 per harness convention
  const float* W1  = (const float*)d_in[2]; const float* b1  = (const float*)d_in[3];
  const float* W2  = (const float*)d_in[4]; const float* b2  = (const float*)d_in[5];
  const float* Wm1 = (const float*)d_in[6]; const float* bm1 = (const float*)d_in[7];
  const float* Wm2 = (const float*)d_in[8]; const float* bm2 = (const float*)d_in[9];

  const int DIN = 512, H0 = 512, H1 = 256, H2 = 256, DOUT = 40, DOUTP = 48;
  const int N = in_sizes[0] / DIN;          // 50000
  const int E = in_sizes[1] / 2;            // 1,600,000
  const int* src = edges;
  const int* dst = edges + E;

  // workspace carve-out (~257 MB)
  char* ws = (char*)d_ws;
  size_t off = 0;
  auto take = [&](size_t bytes) -> char* {
    char* p = ws + off;
    off = (off + bytes + 255) & ~(size_t)255;
    return p;
  };
  float*    dinv = (float*)   take((size_t)N * 4);            // deg -> dinv
  _Float16* Xh   = (_Float16*)take((size_t)N * 512 * 2);      // f16 activations
  _Float16* Wt   = (_Float16*)take((size_t)512 * 512 * 2);    // f16 transposed weight
  float*    Hf   = (float*)   take((size_t)N * 512 * 4);      // GEMM output (f32)
  float*    AG   = (float*)   take((size_t)N * 512 * 4);      // aggregation buffer
  _Float16* Yh   = (_Float16*)AG;                             // reuse as f16 MLP intermediate

  const int TPB = 256;
  const dim3 gsElem(4096);
  const int mblk = (N + 127) / 128;   // 4 waves x 32 rows per block

  // ---- degrees: deg[i] = 1 (self loop) + in-degree; dinv = rsqrt(max(deg,1))
  k_fill     <<<dim3((N + TPB - 1) / TPB), TPB, 0, stream>>>(dinv, 1.0f, N);
  k_deg_edges<<<dim3((E + TPB - 1) / TPB), TPB, 0, stream>>>(dst, dinv, E);
  k_dinv     <<<dim3((N + TPB - 1) / TPB), TPB, 0, stream>>>(dinv, N);

  // ---- layer 1: H = X @ W1 ; agg ; sigmoid
  k_f32_to_f16 <<<gsElem, TPB, 0, stream>>>(features, Xh, (size_t)N * DIN);
  k_transpose_w<<<dim3(1024), TPB, 0, stream>>>(W1, Wt, DIN, H0, H0);
  k_gemm_wmma<2, 2><<<dim3(mblk, H0 / 32), 128, 0, stream>>>(
      Xh, Wt, Hf, nullptr, nullptr, N, DIN, H0, H0, 0);
  k_agg_init <<<gsElem, TPB, 0, stream>>>(Hf, dinv, b1, AG, (size_t)N * H0, H0);
  k_agg_edges<<<dim3(E), H0 / 4, 0, stream>>>(src, dst, dinv, Hf, AG);
  k_sigmoid_f16<<<gsElem, TPB, 0, stream>>>(AG, Xh, (size_t)N * H0);

  // ---- layer 2: H = x1 @ W2 ; agg ; sigmoid
  k_transpose_w<<<dim3(1024), TPB, 0, stream>>>(W2, Wt, H0, H1, H1);
  k_gemm_wmma<2, 2><<<dim3(mblk, H1 / 32), 128, 0, stream>>>(
      Xh, Wt, Hf, nullptr, nullptr, N, H0, H1, H1, 0);
  k_agg_init <<<gsElem, TPB, 0, stream>>>(Hf, dinv, b2, AG, (size_t)N * H1, H1);
  k_agg_edges<<<dim3(E), H1 / 4, 0, stream>>>(src, dst, dinv, Hf, AG);
  k_sigmoid_f16<<<gsElem, TPB, 0, stream>>>(AG, Xh, (size_t)N * H1);

  // ---- MLP head: y = relu(x2 @ Wm1 + bm1) -> f16 ; out = y @ Wm2 + bm2
  k_transpose_w<<<dim3(1024), TPB, 0, stream>>>(Wm1, Wt, H1, H2, H2);
  k_gemm_wmma<2, 2><<<dim3(mblk, H2 / 32), 128, 0, stream>>>(
      Xh, Wt, nullptr, Yh, bm1, N, H1, H2, H2, 1);

  k_transpose_w<<<dim3(64), TPB, 0, stream>>>(Wm2, Wt, H2, DOUT, DOUTP);  // zero-padded to 48 cols
  k_gemm_wmma<2, 3><<<dim3(mblk, 1), 128, 0, stream>>>(
      Yh, Wt, (float*)d_out, nullptr, bm2, N, H2, DOUT, DOUT, 2);
}
